// CtcBoundaryLossV3_90297392431840
// MI455X (gfx1250) — compile-verified, matrix operand order checked by reading, not run on hardware
//
#include <hip/hip_runtime.h>

// CTC boundary loss, collapsed to per-row scalar reductions.
// B=16, T=2048, V=1024, U=256, BLANK=0, LOG_THR=log(3).
//
// Kernel 1: grid (B, S) x 256 threads. Each thread handles one t:
//   - alpha / mask: coalesced f32 loads
//   - blank channel: 4KB-strided gather (one cacheline per lane)
//   - spike count: exact integer count via V_WMMA_I32_16X16X64_IU8
//     (0/1 flag bytes as A-matrix, all-ones B-matrix -> row sums = counts)
//   - alpha partial: f32 wave shuffle reduction
//   Partials written to deterministic ws slots (no atomics, no pre-zero).
// Kernel 2: 1 wave, closed-form epilogue + final scalar.

typedef int v8i __attribute__((ext_vector_type(8)));

#define B_   16
#define T_   2048
#define V_   1024
#define S_   8      // T splits per batch row
#define TPB  256    // 8 wave32 per block; TPB*S_ == T_

__device__ __forceinline__ float wave_sum_f32(float v) {
#pragma unroll
    for (int off = 16; off >= 1; off >>= 1)
        v += __shfl_xor(v, off, 32);
    return v;
}

__global__ void __launch_bounds__(TPB)
ctc_boundary_reduce_kernel(const float* __restrict__ alpha,
                           const float* __restrict__ ctc,
                           const float* __restrict__ mask,
                           float* __restrict__ ws_sum,
                           int*   __restrict__ ws_cnt) {
    const int b   = blockIdx.x;
    const int s   = blockIdx.y;
    const int tid = threadIdx.x;
    const int t   = s * TPB + tid;
    const size_t bt = (size_t)b * T_ + t;

    const float a   = alpha[bt];
    const float m   = mask[bt];
    const float blp = ctc[bt * (size_t)V_];   // BLANK channel, stride V

    const float LOG_THR = 1.0986122886681098f;   // log(3)
    const int flag = (((1.0f - blp) > LOG_THR) && (m != 0.0f)) ? 1 : 0;

    // ---- exact spike count via IU8 WMMA (wave-wide, 32 flags/wave) ----
    // A (16x64 u8): one flag byte per lane in K-byte 0; rest zero.
    // B (64x16 u8): all ones -> D[m][n] = row-count of A.
    v8i A = {flag, 0, 0, 0, 0, 0, 0, 0};
    v8i Bones = {0x01010101, 0x01010101, 0x01010101, 0x01010101,
                 0x01010101, 0x01010101, 0x01010101, 0x01010101};
    v8i C = {0, 0, 0, 0, 0, 0, 0, 0};
    v8i D = __builtin_amdgcn_wmma_i32_16x16x64_iu8(false, A, false, Bones,
                                                   C, false, false);
    int cnt = 0;
#pragma unroll
    for (int i = 0; i < 8; ++i) cnt += D[i];      // rows M=0..7 (lanes<16) / 8..15
    cnt += __shfl_xor(cnt, 16, 32);               // all 16 rows -> wave total

    // ---- f32 alpha partial (exactness matters; keep in f32) ----
    const float wsum = wave_sum_f32(a);

    __shared__ float lsum[TPB / 32];
    __shared__ int   lcnt[TPB / 32];
    const int w = tid >> 5;
    if ((tid & 31) == 0) { lsum[w] = wsum; lcnt[w] = cnt; }
    __syncthreads();
    if (tid == 0) {
        float ssum = 0.0f;
        int   scnt = 0;
#pragma unroll
        for (int i = 0; i < TPB / 32; ++i) { ssum += lsum[i]; scnt += lcnt[i]; }
        ws_sum[b * S_ + s] = ssum;
        ws_cnt[b * S_ + s] = scnt;
    }
}

__global__ void __launch_bounds__(32)
ctc_boundary_finalize_kernel(const float* __restrict__ ws_sum,
                             const int*   __restrict__ ws_cnt,
                             const int*   __restrict__ tlen_raw,
                             float*       __restrict__ out) {
    const int lane = threadIdx.x;
    const bool valid = lane < B_;

    float rsum = 0.0f;
    int   nseg = 0;
    if (valid) {
#pragma unroll
        for (int i = 0; i < S_; ++i) {
            rsum += ws_sum[lane * S_ + i];
            nseg += ws_cnt[lane * S_ + i];
        }
    }

    // text_length: values in [1,256]. If stored as int64 (LE), word[1]==0;
    // if int32, word[1]>=1. Detect element stride accordingly.
    const int stride = (tlen_raw[1] == 0) ? 2 : 1;
    const int tl = valid ? tlen_raw[lane * stride] : 0;

    const bool has  = nseg >= 1;
    const int  leni = valid ? (has ? nseg : 1) : 0;

    int m1 = leni, m2 = tl;
#pragma unroll
    for (int off = 16; off >= 1; off >>= 1) {
        m1 = max(m1, __shfl_xor(m1, off, 32));
        m2 = max(m2, __shfl_xor(m2, off, 32));
    }
    const int L = min(m1, m2);

    float per = 0.0f;
    if (valid) {
        const int M = min(tl, L);               // columns surviving text_mask
        if (has) {
            const int c1 = min(nseg, M);        // cols with |row_sum - 1|
            per = (float)c1 * fabsf(rsum - 1.0f) + (float)(M - c1);
        } else {
            per = (float)(M > 0 ? M - 1 : 0);   // j==0 contributes 0, rest 1
        }
    }
    per = wave_sum_f32(per);
    if (lane == 0) out[0] = per / (float)B_;
}

extern "C" void kernel_launch(void* const* d_in, const int* in_sizes, int n_in,
                              void* d_out, int out_size, void* d_ws, size_t ws_size,
                              hipStream_t stream) {
    const float* alpha = (const float*)d_in[0];
    const float* ctc   = (const float*)d_in[1];
    const float* mask  = (const float*)d_in[2];
    const int*   tlen  = (const int*)d_in[3];

    float* ws_sum = (float*)d_ws;
    int*   ws_cnt = (int*)((char*)d_ws + B_ * S_ * sizeof(float));

    dim3 grid(B_, S_);
    ctc_boundary_reduce_kernel<<<grid, TPB, 0, stream>>>(alpha, ctc, mask,
                                                         ws_sum, ws_cnt);
    ctc_boundary_finalize_kernel<<<1, 32, 0, stream>>>(ws_sum, ws_cnt, tlen,
                                                       (float*)d_out);
}
